// SimpleTransformer_40999757807622
// MI455X (gfx1250) — matrix operand order repaired and checked by reading, not compiled, not run on hardware
//
#include <hip/hip_runtime.h>

// ---------------- problem constants ----------------
#define BATCH   8
#define NSEQ    8192
#define BN      65536        // BATCH*NSEQ
#define DMODEL  256
#define HEADS   4
#define DKH     64
#define PADP    72           // (DKH + POS=2) padded to 72
#define HP      288          // HEADS*PADP
#define DFFN    1024
#define LAYERS  4
#define KVCH    8            // NSEQ/1024 reduction chunks
#define KVSZ    5184         // 72*72

typedef __attribute__((ext_vector_type(16))) __bf16       bf16x16;
typedef __attribute__((ext_vector_type(8)))  float        f32x8;
typedef __attribute__((ext_vector_type(8)))  unsigned int u32x8;

union Frag { u32x8 u; bf16x16 b; };

__device__ __forceinline__ float bf2f(unsigned short h) {
  unsigned int v = ((unsigned int)h) << 16;
  return __builtin_bit_cast(float, v);
}
__device__ __forceinline__ unsigned short f2bf(float f) {
  unsigned int x = __builtin_bit_cast(unsigned int, f);
  x += 0x7FFFu + ((x >> 16) & 1u);          // round-to-nearest-even
  return (unsigned short)(x >> 16);
}
__device__ __forceinline__ float siluf(float v) { return v / (1.f + __expf(-v)); }

// ---------------- generic bf16 WMMA GEMM ----------------
// C[64 x 256 tile] = A[M x K](bf16, LDS-staged) * Bt[N x K](bf16, pre-transposed,
// fragment-ready, loaded straight from global/L2) + epilogue
#define EPI_QKV    0   // +bias -> bf16 into per-head padded layout [BN, HP]
#define EPI_SILU   1   // +bias, silu -> bf16 [BN, outLd]
#define EPI_RES_F  2   // +bias +resid -> f32 [BN, outLd]
#define EPI_RES_FB 3   // +bias +resid -> f32 and bf16

#define SAS 40   // sA row stride (elems): 80B pitch, 16B aligned, conflict-free

template<int EPI>
__global__ __launch_bounds__(256)
void gemm_k(const unsigned short* __restrict__ A, int K,
            const unsigned short* __restrict__ Bt,   // [Ntot][K] bf16
            const float* __restrict__ bias,
            const float* __restrict__ resid,
            float* __restrict__ outF,
            unsigned short* __restrict__ outB,
            int outLd)
{
  __shared__ unsigned short sA[2][64 * SAS];    // double-buffered A tiles only

  const int tid  = threadIdx.x;
  const int lane = tid & 31, wid = tid >> 5;
  const int wm = wid & 3, wn = wid >> 2;        // 4x2 wave grid over 64x256 tile
  const int l16 = lane & 15, lh = lane >> 4;
  const int mBase = blockIdx.x * 64;
  const int nBase = blockIdx.y * 256;

  f32x8 acc[8] = {};

  // A staging: thread -> (row, 8-elem chunk)
  const int ar = tid >> 2;                      // 0..63
  const int ac = (tid & 3) << 3;                // 0,8,16,24
  const unsigned short* aRow = A + (size_t)(mBase + ar) * K + ac;

  // Per-lane B row base: n = nBase + wn*128 + nt*16 + l16, k-offset lh*16.
  // B k order used here (contiguous, same as A): lanes 0-15 K0..15, lanes 16-31 K16..31.
  const unsigned short* bRow = Bt + (size_t)(nBase + wn * 128 + l16) * K + lh * 16;
  const size_t bNt = (size_t)16 * K;            // n-tile stride

  uint4 aReg = *(const uint4*)(aRow);
  *(uint4*)&sA[0][ar * SAS + ac] = aReg;
  __syncthreads();

  const int nK = K >> 5;
  for (int kb = 0; kb < nK; ++kb) {
    const int cur = kb & 1;
    if (kb + 1 < nK) aReg = *(const uint4*)(aRow + ((kb + 1) << 5));
    if (kb + 2 < nK) __builtin_prefetch(aRow + ((kb + 2) << 5), 0, 3);  // WGP-scope

    // Prime the B-fragment pipeline (global loads, 2x16B per fragment)
    Frag pipe[2];
    {
      const unsigned short* bp = bRow + (kb << 5);
      uint4 b0 = *(const uint4*)(bp);
      uint4 b1 = *(const uint4*)(bp + 8);
      pipe[0].u[0] = b0.x; pipe[0].u[1] = b0.y; pipe[0].u[2] = b0.z; pipe[0].u[3] = b0.w;
      pipe[0].u[4] = b1.x; pipe[0].u[5] = b1.y; pipe[0].u[6] = b1.z; pipe[0].u[7] = b1.w;
    }
    // A fragment: two contiguous 16B LDS loads
    Frag af;
    {
      const unsigned short* ap = &sA[cur][(wm * 16 + l16) * SAS + lh * 16];
      uint4 a0 = *(const uint4*)(ap);
      uint4 a1 = *(const uint4*)(ap + 8);
      af.u[0] = a0.x; af.u[1] = a0.y; af.u[2] = a0.z; af.u[3] = a0.w;
      af.u[4] = a1.x; af.u[5] = a1.y; af.u[6] = a1.z; af.u[7] = a1.w;
    }
    // 1-stage pipelined: issue loads for fragment nt+1 before WMMA on fragment nt
#pragma unroll
    for (int nt = 0; nt < 8; ++nt) {
      if (nt + 1 < 8) {
        const unsigned short* bp = bRow + (size_t)(nt + 1) * bNt + (kb << 5);
        uint4 b0 = *(const uint4*)(bp);
        uint4 b1 = *(const uint4*)(bp + 8);
        Frag& nx = pipe[(nt + 1) & 1];
        nx.u[0] = b0.x; nx.u[1] = b0.y; nx.u[2] = b0.z; nx.u[3] = b0.w;
        nx.u[4] = b1.x; nx.u[5] = b1.y; nx.u[6] = b1.z; nx.u[7] = b1.w;
      }
      acc[nt] = __builtin_amdgcn_wmma_f32_16x16x32_bf16(
          false, af.b, false, pipe[nt & 1].b, (short)0, acc[nt], false, false);
    }

    if (kb + 1 < nK) *(uint4*)&sA[cur ^ 1][ar * SAS + ac] = aReg;
    __syncthreads();
  }

  // epilogue: acc[nt][e] -> row = wm*16 + lh*8 + e, col = wn*128 + nt*16 + l16
#pragma unroll
  for (int nt = 0; nt < 8; ++nt) {
#pragma unroll
    for (int e = 0; e < 8; ++e) {
      int r = mBase + wm * 16 + lh * 8 + e;
      int c = nBase + wn * 128 + nt * 16 + l16;
      float v = acc[nt][e] + bias[c];
      if (EPI == EPI_QKV) {
        int dc = (c >> 6) * PADP + 2 + (c & 63);     // head-padded slot (pos at 0..1)
        outB[(size_t)r * HP + dc] = f2bf(v);
      } else if (EPI == EPI_SILU) {
        outB[(size_t)r * outLd + c] = f2bf(siluf(v));
      } else if (EPI == EPI_RES_F) {
        v += resid[(size_t)r * outLd + c];
        outF[(size_t)r * outLd + c] = v;
      } else {
        v += resid[(size_t)r * outLd + c];
        outF[(size_t)r * outLd + c] = v;
        outB[(size_t)r * outLd + c] = f2bf(v);
      }
    }
  }
}

// -------- fixup: write pos into cols 0..1, zero cols 66..71, optional per-head LN --------
__global__ __launch_bounds__(256)
void fixup_qkv(unsigned short* __restrict__ buf,
               const float* __restrict__ gamma, const float* __restrict__ beta,
               const float* __restrict__ pos, int doLN)
{
  int wg   = blockIdx.x * 8 + (threadIdx.x >> 5);   // wave id = token*HEADS + h
  int lane = threadIdx.x & 31;
  int row  = wg >> 2;
  int h    = wg & 3;
  unsigned short* p = buf + (size_t)row * HP + h * PADP;
  if (lane < 2) p[lane] = f2bf(pos[(size_t)row * 2 + lane]);
  if (lane < 6) p[66 + lane] = 0;
  if (doLN) {
    int j0 = lane * 2;
    float x0 = bf2f(p[2 + j0]);
    float x1 = bf2f(p[3 + j0]);
    float s = x0 + x1, q = x0 * x0 + x1 * x1;
    for (int off = 16; off > 0; off >>= 1) {
      s += __shfl_xor(s, off, 32);
      q += __shfl_xor(q, off, 32);
    }
    float mu  = s * (1.f / 64.f);
    float var = q * (1.f / 64.f) - mu * mu;
    float inv = rsqrtf(var + 1e-5f);
    const float* g = gamma + h * DKH;
    const float* b = beta  + h * DKH;
    p[2 + j0] = f2bf((x0 - mu) * inv * g[j0]     + b[j0]);
    p[3 + j0] = f2bf((x1 - mu) * inv * g[j0 + 1] + b[j0 + 1]);
  }
}

// -------- kv partials: kv_part[b,h,chunk][72x72] = sum_{n in chunk} k_n v_n^T / N --------
__global__ __launch_bounds__(256)
void kv_accum(const unsigned short* __restrict__ kp,
              const unsigned short* __restrict__ vp,
              float* __restrict__ part)
{
  int bid   = blockIdx.x;
  int chunk = bid % KVCH;
  int bh    = bid / KVCH;
  int h = bh % HEADS, b = bh / HEADS;
  int tid = threadIdx.x;
  __shared__ float sk[16 * 72], sv[16 * 72];
  float accv[21];
  int dArr[21], eArr[21];
#pragma unroll
  for (int i = 0; i < 21; ++i) {
    accv[i] = 0.f;
    int c = tid + 256 * i;
    if (c < KVSZ) { dArr[i] = c / 72; eArr[i] = c - dArr[i] * 72; } else dArr[i] = -1;
  }
  size_t rowBase = (size_t)b * NSEQ + (size_t)chunk * 1024;
  for (int tb = 0; tb < 64; ++tb) {
#pragma unroll
    for (int i = 0; i < 9; ++i) {
      int flat = tid + 256 * i;
      if (flat < 2304) {
        int tok = flat / 144; int idx = flat - tok * 144;
        size_t r = (rowBase + tb * 16 + tok) * (size_t)HP + h * PADP;
        if (idx < 72) sk[tok * 72 + idx] = bf2f(kp[r + idx]);
        else          sv[tok * 72 + (idx - 72)] = bf2f(vp[r + (idx - 72)]);
      }
    }
    __syncthreads();
#pragma unroll
    for (int i = 0; i < 21; ++i) {
      if (dArr[i] >= 0) {
        float a = 0.f;
#pragma unroll
        for (int t = 0; t < 16; ++t) a += sk[t * 72 + dArr[i]] * sv[t * 72 + eArr[i]];
        accv[i] += a;
      }
    }
    __syncthreads();
  }
  const float scale = 1.f / (float)NSEQ;
#pragma unroll
  for (int i = 0; i < 21; ++i)
    if (dArr[i] >= 0) part[(size_t)bid * KVSZ + tid + 256 * i] = accv[i] * scale;
}

__global__ void kv_reduce(const float* __restrict__ part, float* __restrict__ kv)
{
  int i = blockIdx.x * 256 + threadIdx.x;
  if (i >= BATCH * HEADS * KVSZ) return;
  int bh = i / KVSZ; int c = i - bh * KVSZ;
  float s = 0.f;
  for (int ch = 0; ch < KVCH; ++ch) s += part[(size_t)(bh * KVCH + ch) * KVSZ + c];
  kv[i] = s;
}

// -------- att[n, e] = sum_d q[n, d] * kv[d, e], kv cached in LDS --------
__global__ __launch_bounds__(256)
void att_k(const unsigned short* __restrict__ qp,
           const float* __restrict__ kv,
           unsigned short* __restrict__ att)
{
  const int CH = NSEQ / 256;
  int bid = blockIdx.x;
  int chunk = bid % CH;
  int bh = bid / CH;
  int h = bh % HEADS, b = bh / HEADS;
  __shared__ float skv[KVSZ];
  for (int i = threadIdx.x; i < KVSZ; i += 256) skv[i] = kv[(size_t)bh * KVSZ + i];
  __syncthreads();
  int row = b * NSEQ + chunk * 256 + threadIdx.x;
  const unsigned short* qr = qp + (size_t)row * HP + h * PADP;
  float q[72];
  {
    const uint4* q4 = (const uint4*)qr;           // 16B-aligned: (row*288+h*72)*2
#pragma unroll
    for (int j = 0; j < 9; ++j) {
      uint4 u = q4[j];
      const unsigned short* s = (const unsigned short*)&u;
#pragma unroll
      for (int t = 0; t < 8; ++t) q[j * 8 + t] = bf2f(s[t]);
    }
  }
  unsigned short* ar = att + (size_t)row * HP + h * PADP;
  for (int e = 0; e < 72; ++e) {
    float s = 0.f;
#pragma unroll
    for (int d = 0; d < 72; ++d) s += q[d] * skv[d * 72 + e];   // LDS broadcast
    ar[e] = f2bf(s);
  }
}

// -------- row LayerNorm over 256, wave-per-row, writes f32 + bf16 mirror --------
__global__ __launch_bounds__(256)
void ln_rows(float* __restrict__ x, unsigned short* __restrict__ xb,
             const float* __restrict__ g, const float* __restrict__ bta)
{
  int row  = blockIdx.x * 8 + (threadIdx.x >> 5);
  int lane = threadIdx.x & 31;
  float* xr = x + (size_t)row * DMODEL;
  float v[8];
  float s = 0.f, q = 0.f;
#pragma unroll
  for (int i = 0; i < 8; ++i) { v[i] = xr[lane * 8 + i]; s += v[i]; q += v[i] * v[i]; }
  for (int off = 16; off > 0; off >>= 1) { s += __shfl_xor(s, off, 32); q += __shfl_xor(q, off, 32); }
  float mu  = s * (1.f / 256.f);
  float var = q * (1.f / 256.f) - mu * mu;
  float inv = rsqrtf(var + 1e-5f);
#pragma unroll
  for (int i = 0; i < 8; ++i) {
    int c = lane * 8 + i;
    float y = (v[i] - mu) * inv * g[c] + bta[c];
    xr[c] = y;
    xb[(size_t)row * DMODEL + c] = f2bf(y);
  }
}

// -------- x = node @ Wf + bf (K=3), writes f32 + bf16 --------
__global__ void feat_k(const float* __restrict__ node, const float* __restrict__ Wf,
                       const float* __restrict__ bfv, float* __restrict__ x,
                       unsigned short* __restrict__ xb)
{
  size_t i = (size_t)blockIdx.x * 256 + threadIdx.x;
  int d = (int)(i & 255); size_t bn = i >> 8;
  float s = bfv[d];
#pragma unroll
  for (int f = 0; f < 3; ++f) s += node[bn * 3 + f] * Wf[f * 256 + d];
  x[i] = s; xb[i] = f2bf(s);
}

// -------- weight transpose+convert: src [K][N] f32 -> dst [N][K] bf16 --------
__global__ void wt_k(const float* __restrict__ src, unsigned short* __restrict__ dst,
                     int K, int N)
{
  int i = blockIdx.x * 256 + threadIdx.x;
  if (i >= K * N) return;
  int n = i / K, k = i - n * K;
  dst[i] = f2bf(src[(size_t)k * N + n]);
}

// Wo [L,264,256] -> transposed padded bf16 [L][256][288] (zero cols for pad slots)
__global__ void wot_k(const float* __restrict__ Wo, unsigned short* __restrict__ dst)
{
  int i = blockIdx.x * 256 + threadIdx.x;           // L*256*288
  if (i >= LAYERS * DMODEL * HP) return;
  int l = i / (DMODEL * HP); int rem = i - l * (DMODEL * HP);
  int n = rem / HP; int k = rem - n * HP;
  int h = k / PADP, j = k - h * PADP;
  float v = (j < 66) ? Wo[((size_t)l * 264 + h * 66 + j) * DMODEL + n] : 0.f;
  dst[i] = f2bf(v);
}

// -------- head: out[row] = r2[row,:] . Wout + bout --------
__global__ void head_out(const unsigned short* __restrict__ r2,
                         const float* __restrict__ Wout, const float* __restrict__ bout,
                         float* __restrict__ out)
{
  int row = blockIdx.x * 256 + threadIdx.x;
  float s = bout[0];
  const uint4* r4 = (const uint4*)(r2 + (size_t)row * DMODEL);   // 512B row, aligned
  for (int j = 0; j < 32; ++j) {
    uint4 u = r4[j];
    const unsigned short* sp = (const unsigned short*)&u;
#pragma unroll
    for (int t = 0; t < 8; ++t) s += bf2f(sp[t]) * Wout[j * 8 + t];
  }
  out[row] = s;
}

// ---------------- host launcher ----------------
extern "C" void kernel_launch(void* const* d_in, const int* in_sizes, int n_in,
                              void* d_out, int out_size, void* d_ws, size_t ws_size,
                              hipStream_t stream)
{
  const float* node = (const float*)d_in[0];
  const float* pos  = (const float*)d_in[1];
  const float* Wf   = (const float*)d_in[2];
  const float* bfv  = (const float*)d_in[3];
  const float* Wq   = (const float*)d_in[4];
  const float* Wk   = (const float*)d_in[5];
  const float* Wv   = (const float*)d_in[6];
  const float* bq   = (const float*)d_in[7];
  const float* bk   = (const float*)d_in[8];
  const float* bv   = (const float*)d_in[9];
  const float* lnkg = (const float*)d_in[10];
  const float* lnkb = (const float*)d_in[11];
  const float* lnvg = (const float*)d_in[12];
  const float* lnvb = (const float*)d_in[13];
  const float* Wo   = (const float*)d_in[14];
  const float* bo   = (const float*)d_in[15];
  const float* ln1g = (const float*)d_in[16];
  const float* ln1b = (const float*)d_in[17];
  const float* W1   = (const float*)d_in[18];
  const float* b1   = (const float*)d_in[19];
  const float* W2   = (const float*)d_in[20];
  const float* b2   = (const float*)d_in[21];
  const float* Wr1  = (const float*)d_in[22];
  const float* br1  = (const float*)d_in[23];
  const float* Wr2  = (const float*)d_in[24];
  const float* br2  = (const float*)d_in[25];
  const float* Wout = (const float*)d_in[26];
  const float* bout = (const float*)d_in[27];
  float* out = (float*)d_out;

  char* ws = (char*)d_ws;
  size_t off = 0;
  auto alloc = [&](size_t bytes) {
    void* p = ws + off; off += (bytes + 255) & ~(size_t)255; return p;
  };
  float*          x   = (float*)alloc((size_t)BN * DMODEL * 4);
  unsigned short* xb  = (unsigned short*)alloc((size_t)BN * DMODEL * 2);
  unsigned short* qp  = (unsigned short*)alloc((size_t)BN * HP * 2);
  unsigned short* kp  = (unsigned short*)alloc((size_t)BN * HP * 2);
  unsigned short* vp  = (unsigned short*)alloc((size_t)BN * HP * 2);
  unsigned short* at  = (unsigned short*)alloc((size_t)BN * HP * 2);
  float*          kvP = (float*)alloc((size_t)BATCH * HEADS * KVCH * KVSZ * 4);
  float*          kv  = (float*)alloc((size_t)BATCH * HEADS * KVSZ * 4);
  unsigned short* WqT = (unsigned short*)alloc((size_t)LAYERS * DMODEL * DMODEL * 2);
  unsigned short* WkT = (unsigned short*)alloc((size_t)LAYERS * DMODEL * DMODEL * 2);
  unsigned short* WvT = (unsigned short*)alloc((size_t)LAYERS * DMODEL * DMODEL * 2);
  unsigned short* WoT = (unsigned short*)alloc((size_t)LAYERS * DMODEL * HP * 2);
  unsigned short* W1T = (unsigned short*)alloc((size_t)LAYERS * DMODEL * DFFN * 2);
  unsigned short* W2T = (unsigned short*)alloc((size_t)LAYERS * DFFN * DMODEL * 2);
  unsigned short* Wr1T = (unsigned short*)alloc((size_t)DMODEL * DMODEL * 2);
  unsigned short* Wr2T = (unsigned short*)alloc((size_t)DMODEL * DMODEL * 2);
  // FFN intermediate (BN x 1024 bf16 = 128MB) aliases the dead q/k/v/att region (151MB)
  unsigned short* hb  = qp;

  // ---- weight transpose+convert to fragment-ready [N][K] bf16 ----
  const int nDD = DMODEL * DMODEL;         // 65536
  const int nDF = DMODEL * DFFN;           // 262144
  for (int l = 0; l < LAYERS; ++l) {
    wt_k<<<(nDD + 255) / 256, 256, 0, stream>>>(Wq + (size_t)l * nDD, WqT + (size_t)l * nDD, DMODEL, DMODEL);
    wt_k<<<(nDD + 255) / 256, 256, 0, stream>>>(Wk + (size_t)l * nDD, WkT + (size_t)l * nDD, DMODEL, DMODEL);
    wt_k<<<(nDD + 255) / 256, 256, 0, stream>>>(Wv + (size_t)l * nDD, WvT + (size_t)l * nDD, DMODEL, DMODEL);
    wt_k<<<(nDF + 255) / 256, 256, 0, stream>>>(W1 + (size_t)l * nDF, W1T + (size_t)l * nDF, DMODEL, DFFN);
    wt_k<<<(nDF + 255) / 256, 256, 0, stream>>>(W2 + (size_t)l * nDF, W2T + (size_t)l * nDF, DFFN, DMODEL);
  }
  wt_k<<<(nDD + 255) / 256, 256, 0, stream>>>(Wr1, Wr1T, DMODEL, DMODEL);
  wt_k<<<(nDD + 255) / 256, 256, 0, stream>>>(Wr2, Wr2T, DMODEL, DMODEL);
  int nWo = LAYERS * DMODEL * HP;
  wot_k<<<(nWo + 255) / 256, 256, 0, stream>>>(Wo, WoT);

  // ---- feature transform ----
  feat_k<<<BN * DMODEL / 256, 256, 0, stream>>>(node, Wf, bfv, x, xb);

  const dim3 g1(BN / 64, 1), gF(BN / 64, DFFN / 256);
  for (int l = 0; l < LAYERS; ++l) {
    // QKV projections into padded per-head layout
    gemm_k<EPI_QKV><<<g1, 256, 0, stream>>>(xb, DMODEL, WqT + (size_t)l * nDD,
                                            bq + l * DMODEL, nullptr, nullptr, qp, HP);
    gemm_k<EPI_QKV><<<g1, 256, 0, stream>>>(xb, DMODEL, WkT + (size_t)l * nDD,
                                            bk + l * DMODEL, nullptr, nullptr, kp, HP);
    gemm_k<EPI_QKV><<<g1, 256, 0, stream>>>(xb, DMODEL, WvT + (size_t)l * nDD,
                                            bv + l * DMODEL, nullptr, nullptr, vp, HP);
    // pos concat + pad zeros (+ per-head LN for k, v)
    fixup_qkv<<<BN * HEADS / 8, 256, 0, stream>>>(qp, nullptr, nullptr, pos, 0);
    fixup_qkv<<<BN * HEADS / 8, 256, 0, stream>>>(kp, lnkg + l * HEADS * DKH, lnkb + l * HEADS * DKH, pos, 1);
    fixup_qkv<<<BN * HEADS / 8, 256, 0, stream>>>(vp, lnvg + l * HEADS * DKH, lnvb + l * HEADS * DKH, pos, 1);
    // kv = k^T v / N (deterministic chunked reduction)
    kv_accum<<<BATCH * HEADS * KVCH, 256, 0, stream>>>(kp, vp, kvP);
    kv_reduce<<<(BATCH * HEADS * KVSZ + 255) / 256, 256, 0, stream>>>(kvP, kv);
    // att = q @ kv
    att_k<<<BATCH * HEADS * (NSEQ / 256), 256, 0, stream>>>(qp, kv, at);
    // x = x + att @ Wo + bo  (in-place f32)
    gemm_k<EPI_RES_F><<<g1, 256, 0, stream>>>(at, HP, WoT + (size_t)l * DMODEL * HP,
                                              bo + l * DMODEL, x, x, nullptr, DMODEL);
    // x = LN(x); refresh bf16 mirror
    ln_rows<<<BN / 8, 256, 0, stream>>>(x, xb, ln1g + l * DMODEL, ln1b + l * DMODEL);
    // FFN: h = silu(x @ W1 + b1); x = x + h @ W2 + b2
    gemm_k<EPI_SILU><<<gF, 256, 0, stream>>>(xb, DMODEL, W1T + (size_t)l * nDF,
                                             b1 + l * DFFN, nullptr, nullptr, hb, DFFN);
    gemm_k<EPI_RES_FB><<<g1, 256, 0, stream>>>(hb, DFFN, W2T + (size_t)l * nDF,
                                               b2 + l * DMODEL, x, x, xb, DMODEL);
  }

  // ---- regressor head ----
  unsigned short* r1b = qp;                          // reuse dead region
  unsigned short* r2b = qp + (size_t)BN * DMODEL;
  gemm_k<EPI_SILU><<<g1, 256, 0, stream>>>(xb, DMODEL, Wr1T, br1, nullptr, nullptr, r1b, DMODEL);
  gemm_k<EPI_SILU><<<g1, 256, 0, stream>>>(r1b, DMODEL, Wr2T, br2, nullptr, nullptr, r2b, DMODEL);
  head_out<<<BN / 256, 256, 0, stream>>>(r2b, Wout, bout, out);

  (void)in_sizes; (void)n_in; (void)out_size; (void)ws_size;
}
